// FuzzyTM_13022340842106
// MI455X (gfx1250) — matrix operand order repaired and checked by reading, not compiled, not run on hardware
//
#include <hip/hip_runtime.h>

// ---------------------------------------------------------------------------
// Fuzzy-TM forward scan for MI455X (gfx1250).
//
// Phase 1a: softmax T rows -> bf16, stored pre-swizzled in the exact
//           V_WMMA_F32_16X16X32_BF16 B-fragment layout (1 tile = 32x16 = 1KB,
//           lane-contiguous so the scan does coalesced b128 loads).
// Phase 1b: softmax O rows -> f32; copy initial state; zero grid barrier.
// Phase 2:  persistent 32-block scan kernel, 4096 steps, one grid barrier per
//           step (monotonic atomic counter). Per step each of the 256 waves
//           owns one 16-column tile and runs 128 bf16 WMMAs over K=4096.
//           Both softmaxed matrices (64MB bf16) stay resident in the 192MB L2.
// ---------------------------------------------------------------------------

#define NSTATES 4096
#define SEQLEN  4096
#define TPB     256
#define NBLK    32
#define WAVES   8          // TPB/32
#define KTILES  128        // NSTATES/32
#define TILE_ELEMS 512     // 32x16 bf16 elements per fragment tile

typedef __attribute__((ext_vector_type(16))) __bf16 v16bf;
typedef __attribute__((ext_vector_type(8)))  __bf16 v8bf;
typedef __attribute__((ext_vector_type(8)))  float  v8f;

union BF16x16 { v16bf v; v8bf h[2]; };

// ------------------------- workspace layout (bytes) ------------------------
constexpr size_t TF_BYTES  = 2ull * NSTATES * NSTATES * 2;       // 64 MB bf16
constexpr size_t OFF_OM    = TF_BYTES;
constexpr size_t OM_BYTES  = 2ull * NSTATES * 4 * sizeof(float); // 128 KB
constexpr size_t OFF_ST    = OFF_OM + OM_BYTES;
constexpr size_t ST_BYTES  = 2ull * NSTATES * sizeof(float);     // 32 KB
constexpr size_t OFF_CTR   = OFF_ST + ST_BYTES;

// ---------------------------------------------------------------------------
// Phase 1a: one block per (k, s) row of T. Softmax over n, emit bf16 into the
// WMMA B-fragment swizzle:
//   tile (kt = k>>5, nt = n>>4), lane = (n&15) + 16*((k>>3)&1),
//   elem = (k&7) | (((k&31)>>4)<<3)
//   elem_index = s*N*N + ((nt*KTILES + kt)*32 + lane)*16 + elem
// ---------------------------------------------------------------------------
__global__ __launch_bounds__(TPB)
void ftm_softmaxT_swizzle(const float* __restrict__ T, __bf16* __restrict__ tf) {
    const int row = blockIdx.x;            // 0..8191  == k*2 + s
    const int k = row >> 1;
    const int s = row & 1;
    const float* __restrict__ r = T + (size_t)row * NSTATES;
    const int tid = threadIdx.x;

    float v[16];
    float mx = -__builtin_inff();
    #pragma unroll
    for (int i = 0; i < 16; ++i) {
        v[i] = r[tid + i * TPB];
        mx = fmaxf(mx, v[i]);
    }

    __shared__ float red[TPB];
    red[tid] = mx; __syncthreads();
    for (int off = TPB / 2; off > 0; off >>= 1) {
        if (tid < off) red[tid] = fmaxf(red[tid], red[tid + off]);
        __syncthreads();
    }
    mx = red[0]; __syncthreads();

    float sum = 0.f;
    #pragma unroll
    for (int i = 0; i < 16; ++i) { v[i] = __expf(v[i] - mx); sum += v[i]; }
    red[tid] = sum; __syncthreads();
    for (int off = TPB / 2; off > 0; off >>= 1) {
        if (tid < off) red[tid] += red[tid + off];
        __syncthreads();
    }
    const float inv = 1.0f / red[0];

    const int kt = k >> 5, kl = k & 31;
    #pragma unroll
    for (int i = 0; i < 16; ++i) {
        const int n = tid + i * TPB;
        const int nt = n >> 4, nl = n & 15;
        const size_t idx = (size_t)s * NSTATES * NSTATES +
                           (((size_t)nt * KTILES + kt) * 32 + nl + ((kl >> 3) & 1) * 16) * 16 + ((kl & 7) | ((kl >> 4) << 3));
        tf[idx] = (__bf16)(v[i] * inv);
    }
}

// ---------------------------------------------------------------------------
// Phase 1b: softmax O rows (length 4), copy initial state, zero barrier ctr.
// ---------------------------------------------------------------------------
__global__ __launch_bounds__(TPB)
void ftm_softmaxO_init(const float* __restrict__ O,
                       const float* __restrict__ state0,
                       float* __restrict__ om, float* __restrict__ st,
                       unsigned* __restrict__ ctr) {
    const int i = blockIdx.x * blockDim.x + threadIdx.x;  // 0..8191
    if (i < 2 * NSTATES) {
        const int k = i >> 1, s = i & 1;
        const float* r = O + (size_t)i * 4;
        float a = r[0], b = r[1], c = r[2], d = r[3];
        const float mx = fmaxf(fmaxf(a, b), fmaxf(c, d));
        a = __expf(a - mx); b = __expf(b - mx);
        c = __expf(c - mx); d = __expf(d - mx);
        const float inv = 1.0f / (a + b + c + d);
        float* dst = om + ((size_t)s * NSTATES + k) * 4;
        dst[0] = a * inv; dst[1] = b * inv; dst[2] = c * inv; dst[3] = d * inv;
    }
    if (i < NSTATES) st[i] = state0[i];
    if (i == 0) *ctr = 0u;
}

// ---------------------------------------------------------------------------
// Phase 2: persistent scan.
// ---------------------------------------------------------------------------
#define WMMA_STEP(ACC, KTI)                                                    \
    {                                                                          \
        const int kb = (KTI) * 32 + khalf;                                     \
        BF16x16 afr, bfr;                                                      \
        afr.h[0] = *(const v8bf*)&sstate[kb];        /* K = k0+half+0..7   */  \
        afr.h[1] = *(const v8bf*)&sstate[kb + 16];   /* K = k0+half+16..23 */  \
        const __bf16* p = tbase + (size_t)(KTI) * TILE_ELEMS + lane * 16;      \
        bfr.h[0] = *(const v8bf*)(p);                                          \
        bfr.h[1] = *(const v8bf*)(p + 8);                                      \
        ACC = __builtin_amdgcn_wmma_f32_16x16x32_bf16(                         \
            false, afr.v, false, bfr.v, (short)0, ACC, false, false);          \
    }

__global__ __launch_bounds__(TPB)
void ftm_scan(const int* __restrict__ seq,
              const __bf16* __restrict__ tf,
              const float* __restrict__ om,
              float* __restrict__ st,
              float* __restrict__ out,
              unsigned* __restrict__ ctr) {
    __shared__ __attribute__((aligned(32))) __bf16 sstate[NSTATES];

    const int tid   = threadIdx.x;
    const int lane  = tid & 31;
    const int wave  = tid >> 5;
    const int nt    = blockIdx.x * WAVES + wave;   // 0..255 column tile
    const int n0    = nt * 16;
    const int khalf = (lane >= 16) ? 8 : 0;        // ISA 16-bit A-matrix layout

    for (int t = 0; t < SEQLEN; ++t) {
        const int cur = t & 1;
        const float* __restrict__ statef = st + (size_t)cur * NSTATES;
        float* __restrict__ nstate       = st + (size_t)(cur ^ 1) * NSTATES;

        // Stage state into LDS as bf16 (DEV-scope loads: state was written by
        // other WGPs last step).
        for (int i = tid; i < NSTATES; i += TPB) {
            const float sv = __hip_atomic_load(&statef[i], __ATOMIC_RELAXED,
                                               __HIP_MEMORY_SCOPE_AGENT);
            sstate[i] = (__bf16)sv;
        }
        __syncthreads();

        const int sym = seq[t] & 1;

        // Emission output: out[t] = state @ Om[:, sym, :]  (block 0, wave 0)
        if (blockIdx.x == 0 && wave == 0) {
            float a0 = 0.f, a1 = 0.f, a2 = 0.f, a3 = 0.f;
            const float* o = om + (size_t)sym * NSTATES * 4;
            for (int k = lane; k < NSTATES; k += 32) {
                const float sv = __hip_atomic_load(&statef[k], __ATOMIC_RELAXED,
                                                   __HIP_MEMORY_SCOPE_AGENT);
                a0 += sv * o[k * 4 + 0];
                a1 += sv * o[k * 4 + 1];
                a2 += sv * o[k * 4 + 2];
                a3 += sv * o[k * 4 + 3];
            }
            for (int off = 16; off > 0; off >>= 1) {
                a0 += __shfl_xor(a0, off);
                a1 += __shfl_xor(a1, off);
                a2 += __shfl_xor(a2, off);
                a3 += __shfl_xor(a3, off);
            }
            if (lane == 0) {
                out[(size_t)t * 4 + 0] = a0;
                out[(size_t)t * 4 + 1] = a1;
                out[(size_t)t * 4 + 2] = a2;
                out[(size_t)t * 4 + 3] = a3;
            }
        }

        // new_state[n0..n0+15] = state @ Tm_sym[:, n0..n0+15]
        const __bf16* tbase = tf + (size_t)sym * NSTATES * NSTATES +
                              (size_t)nt * KTILES * TILE_ELEMS;
        v8f c0 = {}, c1 = {}, c2 = {}, c3 = {};
        for (int kt = 0; kt < KTILES; kt += 4) {
            if (kt + 4 < KTILES)
                __builtin_prefetch(tbase + (size_t)(kt + 4) * TILE_ELEMS + lane * 16);
            WMMA_STEP(c0, kt + 0);
            WMMA_STEP(c1, kt + 1);
            WMMA_STEP(c2, kt + 2);
            WMMA_STEP(c3, kt + 3);
        }
        const v8f cs = (c0 + c1) + (c2 + c3);
        // D layout: VGPR0, lanes 0..15 hold row M=0, N=lane (all rows equal).
        const float nv = cs[0];
        if (lane < 16) {
            __hip_atomic_store(&nstate[n0 + lane], nv, __ATOMIC_RELAXED,
                               __HIP_MEMORY_SCOPE_AGENT);
        }

        // ---- grid-wide barrier (monotonic counter, never reset mid-run) ----
        __threadfence();        // drain this thread's stores to device scope
        __syncthreads();        // whole block arrived + stores issued
        if (tid == 0) {
            __hip_atomic_fetch_add(ctr, 1u, __ATOMIC_ACQ_REL,
                                   __HIP_MEMORY_SCOPE_AGENT);
            const unsigned target = (unsigned)(t + 1) * (unsigned)NBLK;
            while (__hip_atomic_load(ctr, __ATOMIC_ACQUIRE,
                                     __HIP_MEMORY_SCOPE_AGENT) < target) {
                __builtin_amdgcn_s_sleep(1);
            }
        }
        __syncthreads();
    }
}

// ---------------------------------------------------------------------------
extern "C" void kernel_launch(void* const* d_in, const int* in_sizes, int n_in,
                              void* d_out, int out_size, void* d_ws, size_t ws_size,
                              hipStream_t stream) {
    const float* state_dist = (const float*)d_in[0];
    const int*   input_seq  = (const int*)d_in[1];
    const float* T          = (const float*)d_in[2];
    const float* O          = (const float*)d_in[3];
    float*       out        = (float*)d_out;

    char* ws = (char*)d_ws;
    __bf16*   tf  = (__bf16*)ws;                    // 64 MB swizzled bf16 Tm
    float*    om  = (float*)(ws + OFF_OM);          // 128 KB f32 Om
    float*    st  = (float*)(ws + OFF_ST);          // 2 x 4096 f32 state
    unsigned* ctr = (unsigned*)(ws + OFF_CTR);      // grid barrier counter

    ftm_softmaxT_swizzle<<<2 * NSTATES, TPB, 0, stream>>>(T, tf);
    ftm_softmaxO_init<<<(2 * NSTATES + TPB - 1) / TPB, TPB, 0, stream>>>(
        O, state_dist, om, st, ctr);
    ftm_scan<<<NBLK, TPB, 0, stream>>>(input_seq, tf, om, st, out, ctr);
}
